// STGCNVideoPredictor_7404523619180
// MI455X (gfx1250) — compile-verified, matrix operand order checked by reading, not run on hardware
//
#include <hip/hip_runtime.h>
#include <hip/hip_bf16.h>

typedef __attribute__((ext_vector_type(16))) _Float16 v16h;
typedef __attribute__((ext_vector_type(8)))  float    v8f;
typedef __attribute__((ext_vector_type(4)))  unsigned int u32x4;
typedef _Float16 h16;

#define EPSBN 1e-5f
#define MROWS 81920      // B*T*V
#define VV 1024
#define TT 5
#define BB 16

// Branch-free k-segment: nchunks 32-wide K steps over contiguous A and B rows.
__device__ __forceinline__ void gemm_kseg(const h16* __restrict__ arow,
                                          const h16* __restrict__ brow,
                                          long long bstep, int nchunks,
                                          int hofs, v8f (&acc)[4]) {
  for (int kc = 0; kc < nchunks; ++kc) {
    v16h av;
    u32x4* apv = (u32x4*)&av;
    apv[0] = *(const u32x4*)(arow + hofs);
    apv[1] = *(const u32x4*)(arow + 16 + hofs);
    v16h bv[4];
#pragma unroll
    for (int j = 0; j < 4; ++j) {
      const h16* bj = brow + (long long)j * bstep;
      u32x4* bpv = (u32x4*)&bv[j];
      bpv[0] = *(const u32x4*)(bj + hofs);
      bpv[1] = *(const u32x4*)(bj + 16 + hofs);
    }
    __builtin_prefetch(brow + 32, 0, 1);   // gfx1250 global_prefetch_b8
#pragma unroll
    for (int j = 0; j < 4; ++j)
      acc[j] = __builtin_amdgcn_wmma_f32_16x16x32_f16(
          false, av, false, bv[j], (short)0, acc[j], false, false);
    arow += 32;
    brow += 32;
  }
}

// Generic WMMA GEMM: C[M x N](f16) = A(f16) * B^T(f16 [N][K]) + bias(f32)
// mode 0: A row-major [M][Kdim]
// mode 1: implicit temporal im2col over t (pad 1): A = act [b][t][v][Cc],
//         k = dt*Cc + c; V=1024, T=5. trow is wave-uniform (tiles never
//         straddle t), so out-of-range dt segments are skipped scalar-ly.
// mode 2: A = rows of last time step: m=(b,v) -> act[((b*5+4)*1024+v)*Cc]
// All of Kdim % 32 == 0, N % 64 == 0, M % 16 == 0 hold for this network.
// A single gemm_kseg instantiation runs inside one segment loop so the inner
// loop exists exactly once (keeps B tiles in distinct regs, pipelined waits).
__global__ __launch_bounds__(32)
void k_gemm_wmma(const h16* __restrict__ Aact, const h16* __restrict__ Bt,
                 const float* __restrict__ bias, h16* __restrict__ Cout,
                 int M, int N, int Kdim, int mode, int Cc) {
  const int lane = threadIdx.x & 31;
  const int half_ = lane >> 4;
  const int mr = lane & 15;
  const int m0 = blockIdx.x * 16;
  const int n0 = blockIdx.y * 64;
  const int m = m0 + mr;
  const int hofs = half_ * 8;           // element offset of this lane's K-octet

  v8f acc[4];
#pragma unroll
  for (int j = 0; j < 4; ++j) acc[j] = (v8f){0.f,0.f,0.f,0.f,0.f,0.f,0.f,0.f};

  const h16* brow = Bt + (long long)(n0 + mr) * Kdim;  // lane's B row (k = 0)
  const long long bstep = 16LL * Kdim;                 // next 16-col subtile

  const h16* ar;
  const h16* bseg = brow;
  long long astep = 0;
  int bchunk = 0, nseg = 1, nch = Kdim >> 5;
  if (mode == 1) {
    const int trow = (m0 >> 10) % TT;        // wave-uniform time index
    const long long step = (long long)Cc << 10;   // V * Cc
    const int lo = (trow >= 1) ? 0 : 1;
    const int hi = (trow <= TT - 2) ? 2 : 1;
    ar = Aact + (long long)m * Cc + (long long)(lo - 1) * step;
    bseg = brow + lo * Cc;
    astep = step;
    bchunk = Cc;
    nseg = hi - lo + 1;
    nch = Cc >> 5;
  } else if (mode == 2) {
    int b = m >> 10, v = m & 1023;
    ar = Aact + (((long long)(b * TT + (TT - 1)) << 10) + v) * Cc;
  } else {
    ar = Aact + (long long)m * Kdim;
  }

  for (int s = 0; s < nseg; ++s) {
    gemm_kseg(ar, bseg, bstep, nch, hofs, acc);
    ar += astep;
    bseg += bchunk;
  }

  // C/D layout: vgpr i, lanes 0-15 -> M=i, lanes 16-31 -> M=i+8; N = lane%16
#pragma unroll
  for (int j = 0; j < 4; ++j) {
    int col = n0 + 16 * j + mr;
    float bvv = bias ? bias[col] : 0.f;
#pragma unroll
    for (int i = 0; i < 8; ++i) {
      int row = m0 + 8 * half_ + i;
      Cout[(long long)row * N + col] = (h16)(acc[j][i] + bvv);
    }
  }
}

// ---- PatchEmbed im2col: A[m=(b,t,v)][k=(c,p,q)] from x[B,T,3,256,256] ----
__global__ void k_embed_im2col(const float* __restrict__ x, h16* __restrict__ Aout, int total) {
  int idx = blockIdx.x * blockDim.x + threadIdx.x;
  if (idx >= total) return;
  int k = idx % 192, m = idx / 192;
  int c = k / 64, pq = k % 64, p = pq >> 3, q = pq & 7;
  int v = m & 1023, bt = m >> 10;
  int vi = v >> 5, vj = v & 31;
  float val = x[((long long)bt * 3 + c) * 65536 + (long long)(vi * 8 + p) * 256 + (vj * 8 + q)];
  Aout[idx] = (h16)val;
}

__global__ void k_f32_to_f16(const float* __restrict__ in, h16* __restrict__ out, int n) {
  int i = blockIdx.x * blockDim.x + threadIdx.x;
  if (i < n) out[i] = (h16)in[i];
}

// wt [OC][IC][3] -> packed [OC][3*IC] with k = dt*IC + c
__global__ void k_pack_wt(const float* __restrict__ w, h16* __restrict__ out, int OC, int IC) {
  int idx = blockIdx.x * blockDim.x + threadIdx.x;
  int total = OC * IC * 3;
  if (idx >= total) return;
  int o = idx / (3 * IC), rem = idx % (3 * IC);
  int dt = rem / IC, c = rem % IC;
  out[idx] = (h16)w[((long long)o * IC + c) * 3 + dt];
}

// dec_w [256][192] -> packed [pn=192][c=256]
__global__ void k_pack_dec(const float* __restrict__ w, h16* __restrict__ out) {
  int idx = blockIdx.x * blockDim.x + threadIdx.x;
  if (idx >= 192 * 256) return;
  int pn = idx / 256, c = idx % 256;
  out[idx] = (h16)w[(long long)c * 192 + pn];
}

// Sparse graph aggregation: ya[m][c] = yt[m][c] + sum_{v in N(w)} yt[(b,t,v)][OC + c]
__global__ void k_graph_agg(const h16* __restrict__ yt, h16* __restrict__ ya, int OC, int total) {
  int idx = blockIdx.x * blockDim.x + threadIdx.x;
  if (idx >= total) return;
  int c = idx % OC, m = idx / OC;
  int w = m & 1023;
  int i = w >> 5, j = w & 31;
  long long stride = 2LL * OC;
  long long row0 = (long long)(m - w) * stride;     // base of (b,t,v=0)
  float acc = (float)yt[(long long)m * stride + c]; // identity branch (k=0)
  if (i > 0)  acc += (float)yt[row0 + (long long)(w - 32) * stride + OC + c];
  if (i < 31) acc += (float)yt[row0 + (long long)(w + 32) * stride + OC + c];
  if (j > 0)  acc += (float)yt[row0 + (long long)(w - 1) * stride + OC + c];
  if (j < 31) acc += (float)yt[row0 + (long long)(w + 1) * stride + OC + c];
  ya[(long long)m * OC + c] = (h16)acc;
}

// Per-channel batch statistics over M rows (LDS tree reduce)
__global__ __launch_bounds__(256)
void k_bn_stats(const h16* __restrict__ x, float* __restrict__ stats, int M, int C) {
  int c = blockIdx.x;
  float s = 0.f, s2 = 0.f;
  for (int m = threadIdx.x; m < M; m += 256) {
    float v = (float)x[(long long)m * C + c];
    s += v; s2 += v * v;
  }
  __shared__ float sh[2][256];
  sh[0][threadIdx.x] = s; sh[1][threadIdx.x] = s2;
  __syncthreads();
  for (int off = 128; off > 0; off >>= 1) {
    if ((int)threadIdx.x < off) {
      sh[0][threadIdx.x] += sh[0][threadIdx.x + off];
      sh[1][threadIdx.x] += sh[1][threadIdx.x + off];
    }
    __syncthreads();
  }
  if (threadIdx.x == 0) {
    float mean = sh[0][0] / M;
    stats[2 * c]     = mean;
    stats[2 * c + 1] = sh[1][0] / M - mean * mean;
  }
}

__global__ void k_bn_relu(h16* __restrict__ x, const float* __restrict__ st,
                          const float* __restrict__ g, const float* __restrict__ b,
                          int C, int total) {
  int idx = blockIdx.x * blockDim.x + threadIdx.x;
  if (idx >= total) return;
  int c = idx % C;
  float v = ((float)x[idx] - st[2 * c]) * rsqrtf(st[2 * c + 1] + EPSBN) * g[c] + b[c];
  x[idx] = (h16)fmaxf(v, 0.f);
}

__global__ void k_bn_add_relu(const h16* __restrict__ yc, const float* __restrict__ st2,
                              const float* __restrict__ g2, const float* __restrict__ b2,
                              const h16* __restrict__ rr, const float* __restrict__ str,
                              const float* __restrict__ gr, const float* __restrict__ br,
                              const h16* __restrict__ zprev, h16* __restrict__ zout,
                              int C, int total, int has_r) {
  int idx = blockIdx.x * blockDim.x + threadIdx.x;
  if (idx >= total) return;
  int c = idx % C;
  float y = ((float)yc[idx] - st2[2 * c]) * rsqrtf(st2[2 * c + 1] + EPSBN) * g2[c] + b2[c];
  float r;
  if (has_r)
    r = ((float)rr[idx] - str[2 * c]) * rsqrtf(str[2 * c + 1] + EPSBN) * gr[c] + br[c];
  else
    r = (float)zprev[idx];
  zout[idx] = (h16)fmaxf(y + r, 0.f);
}

// Decoder scatter + bias + sigmoid -> f32 out [B,1,3,256,256]
__global__ void k_decode(const h16* __restrict__ y, const float* __restrict__ dec_b,
                         float* __restrict__ out, int total) {
  int idx = blockIdx.x * blockDim.x + threadIdx.x;
  if (idx >= total) return;
  int w = idx & 255, h = (idx >> 8) & 255, d = (idx >> 16) % 3, b = idx / (3 * 65536);
  int v = (h >> 3) * 32 + (w >> 3);
  int n = d * 64 + (h & 7) * 8 + (w & 7);
  float val = (float)y[((long long)b * VV + v) * 192 + n] + dec_b[d];
  out[idx] = 1.f / (1.f + __expf(-val));
}

static inline int cdiv(long long a, int b) { return (int)((a + b - 1) / b); }

extern "C" void kernel_launch(void* const* d_in, const int* in_sizes, int n_in,
                              void* d_out, int out_size, void* d_ws, size_t ws_size,
                              hipStream_t stream) {
  (void)in_sizes; (void)n_in; (void)out_size; (void)ws_size;
  // pytree (sorted-key) flatten order:
  // 0:A | blocks 0..4 | 49:dec_b 50:dec_w 51:embed_b 52:embed_w | 53:x
  struct Blk { int b1,b2,b2b,br,brb,bt,g1,g2,gr,w2,wr,wt,ic,oc,has_r; };
  Blk blks[5] = {
    { 1, 2, 3,-1,-1, 4, 5, 6,-1, 7,-1, 8,  64, 64,0},
    { 9,10,11,-1,-1,12,13,14,-1,15,-1,16,  64, 64,0},
    {17,18,19,20,21,22,23,24,25,26,27,28,  64,128,1},
    {29,30,31,-1,-1,32,33,34,-1,35,-1,36, 128,128,0},
    {37,38,39,40,41,42,43,44,45,46,47,48, 128,256,1},
  };
  const float* dec_b   = (const float*)d_in[49];
  const float* dec_w   = (const float*)d_in[50];
  const float* embed_b = (const float*)d_in[51];
  const float* embed_w = (const float*)d_in[52];
  const float* x       = (const float*)d_in[53];

  char* ws = (char*)d_ws;
  h16* Z0  = (h16*)ws;                                // 42 MB: activation ping
  h16* Z1  = (h16*)(ws + 42ull  * 1024 * 1024);       // 42 MB: activation pong
  h16* SCR = (h16*)(ws + 84ull  * 1024 * 1024);       // 84 MB: im2col / conv outs
  h16* PW  = (h16*)(ws + 168ull * 1024 * 1024);       //  1 MB: packed f16 weights
  float* ST  = (float*)(ws + 169ull * 1024 * 1024);   // stats
  float* ST1 = ST, *ST2 = ST + 512, *STR = ST + 1024;

  const int TPB = 256;

  // ---- PatchEmbed: im2col + GEMM (M=81920, K=192, N=64) ----
  {
    long long tot = (long long)MROWS * 192;
    k_embed_im2col<<<cdiv(tot, TPB), TPB, 0, stream>>>(x, SCR, (int)tot);
    k_f32_to_f16<<<cdiv(12288, TPB), TPB, 0, stream>>>(embed_w, PW, 12288);
    dim3 g(MROWS / 16, 64 / 64);
    k_gemm_wmma<<<g, 32, 0, stream>>>(SCR, PW, embed_b, Z0,
                                      MROWS, 64, 192, 0, 0);
  }

  h16* zc = Z0;  // current activation [m][C]
  h16* zo = Z1;

  for (int bi = 0; bi < 5; ++bi) {
    const Blk& bk = blks[bi];
    int ic = bk.ic, oc = bk.oc, OC2 = oc * 2;
    long long totO = (long long)MROWS * oc;

    // temporal conv 1 (implicit im2col), bias bt, -> SCR [m][2*oc]
    k_pack_wt<<<cdiv((long long)OC2 * ic * 3, TPB), TPB, 0, stream>>>(
        (const float*)d_in[bk.wt], PW, OC2, ic);
    dim3 g1(MROWS / 16, OC2 / 64);
    k_gemm_wmma<<<g1, 32, 0, stream>>>(zc, PW, (const float*)d_in[bk.bt], SCR,
                                       MROWS, OC2, 3 * ic, 1, ic);

    // sparse graph aggregation -> zo [m][oc]
    k_graph_agg<<<cdiv(totO, TPB), TPB, 0, stream>>>(SCR, zo, oc, (int)totO);

    // BN1 + ReLU (in place on zo)
    k_bn_stats<<<oc, 256, 0, stream>>>(zo, ST1, MROWS, oc);
    k_bn_relu<<<cdiv(totO, TPB), TPB, 0, stream>>>(
        zo, ST1, (const float*)d_in[bk.g1], (const float*)d_in[bk.b1], oc, (int)totO);

    // temporal conv 2, bias b2, -> SCR [m][oc]
    k_pack_wt<<<cdiv((long long)oc * oc * 3, TPB), TPB, 0, stream>>>(
        (const float*)d_in[bk.w2], PW, oc, oc);
    dim3 g2(MROWS / 16, oc / 64);
    k_gemm_wmma<<<g2, 32, 0, stream>>>(zo, PW, (const float*)d_in[bk.b2], SCR,
                                       MROWS, oc, 3 * oc, 1, oc);
    k_bn_stats<<<oc, 256, 0, stream>>>(SCR, ST2, MROWS, oc);

    // residual path
    h16* RR = SCR + totO;
    if (bk.has_r) {
      k_f32_to_f16<<<cdiv((long long)oc * ic, TPB), TPB, 0, stream>>>(
          (const float*)d_in[bk.wr], PW, oc * ic);
      dim3 gr(MROWS / 16, oc / 64);
      k_gemm_wmma<<<gr, 32, 0, stream>>>(zc, PW, (const float*)d_in[bk.br], RR,
                                         MROWS, oc, ic, 0, 0);
      k_bn_stats<<<oc, 256, 0, stream>>>(RR, STR, MROWS, oc);
    }

    // fused BN2 + residual + ReLU -> zo (yb no longer needed)
    k_bn_add_relu<<<cdiv(totO, TPB), TPB, 0, stream>>>(
        SCR, ST2, (const float*)d_in[bk.g2], (const float*)d_in[bk.b2b],
        RR, STR,
        bk.has_r ? (const float*)d_in[bk.gr]  : (const float*)d_in[bk.g2],
        bk.has_r ? (const float*)d_in[bk.brb] : (const float*)d_in[bk.b2b],
        zc, zo, oc, (int)totO, bk.has_r);

    h16* t = zc; zc = zo; zo = t;
  }

  // ---- Decoder: GEMM on last time step (M=16384, K=256, N=192) + sigmoid ----
  {
    k_pack_dec<<<cdiv(192 * 256, TPB), TPB, 0, stream>>>(dec_w, PW);
    dim3 gd(16384 / 16, 192 / 64);
    k_gemm_wmma<<<gd, 32, 0, stream>>>(zc, PW, nullptr, SCR,
                                       16384, 192, 256, 2, 256);
    int tot = BB * 3 * 65536;
    k_decode<<<cdiv(tot, TPB), TPB, 0, stream>>>(SCR, dec_b, (float*)d_out, tot);
  }
}
